// HierarchicalGRU_30064771072106
// MI455X (gfx1250) — compile-verified, matrix operand order checked by reading, not run on hardware
//
#include <hip/hip_runtime.h>
#include <hip/hip_bf16.h>
#include <math.h>

// ---------------- types ----------------
typedef __bf16 bf16_t;
typedef __attribute__((ext_vector_type(16))) __bf16 v16bf;
typedef __attribute__((ext_vector_type(8)))  __bf16 v8bf;
typedef __attribute__((ext_vector_type(8)))  float  v8f;

#define NV      50257
#define EDIM    256
#define HDIM    256
#define G3      768         // 3*H
#define BB      8
#define UU      40
#define WW      80
#define NSEQ    (BB*UU)     // 320
#define MWORD   (NSEQ*WW)   // 25600
#define U_OUT_ELEMS  ((size_t)BB*UU*2*HDIM)            // 163840
#define W2_ELEMS     ((size_t)BB*UU*WW*2*HDIM)         // 13107200

// LDS layout for the recurrence kernel:
//   [0, 16K)        : hF   16x256 f32 hidden-state master
//   [16K, 24K)      : hB   16x256 bf16 WMMA operand copy
//   [24K, 24K+256K) : WhL  cached r/z gate rows of Whh (512 x 256 bf16)
#define REC_HF_BYTES   (16 * HDIM * 4)
#define REC_HB_BYTES   (16 * HDIM * 2)
#define REC_WCACHE_ROWS 512
#define REC_WCACHE_BYTES ((size_t)REC_WCACHE_ROWS * HDIM * 2)
#define REC_SMEM_BYTES (REC_HF_BYTES + REC_HB_BYTES + REC_WCACHE_BYTES) // 286720 <= 320K

__device__ __forceinline__ float sigf(float x) { return 1.0f / (1.0f + __expf(-x)); }

// Load a 16x32 bf16 WMMA fragment (A or B operand) from a row-major source.
// Per ISA 7.12.2: lane = (row & 15) + 16*kgroup; per lane the 16 values are
// two contiguous 8-element runs: k = k0 + g*8 + [0..7] and k0 + 16 + g*8 + [0..7].
__device__ __forceinline__ v16bf load_frag(const bf16_t* __restrict__ base, int ld,
                                           int k0, int lane) {
    int r = lane & 15, g = lane >> 4;
    const bf16_t* p = base + (size_t)r * ld + k0 + g * 8;
    v8bf lo = *(const v8bf*)(p);
    v8bf hi = *(const v8bf*)(p + 16);
    v16bf out;
#pragma unroll
    for (int j = 0; j < 8; ++j) { out[j] = lo[j]; out[j + 8] = hi[j]; }
    return out;
}

__device__ __forceinline__ v8f wmma_bf16(v16bf a, v16bf b, v8f c) {
    return __builtin_amdgcn_wmma_f32_16x16x32_bf16(false, a, false, b, (short)0, c,
                                                   false, false);
}

// ---------------- small utility kernels ----------------
__global__ void f32_to_bf16_kernel(const float* __restrict__ src,
                                   bf16_t* __restrict__ dst, int n) {
    for (int i = blockIdx.x * blockDim.x + threadIdx.x; i < n;
         i += gridDim.x * blockDim.x)
        dst[i] = (bf16_t)src[i];
}

__global__ void zero_f32_kernel(float* __restrict__ p, size_t n) {
    for (size_t i = blockIdx.x * (size_t)blockDim.x + threadIdx.x; i < n;
         i += (size_t)gridDim.x * blockDim.x)
        p[i] = 0.0f;
}

__global__ void embed_gather_kernel(const int* __restrict__ ids,
                                    const float* __restrict__ emb,
                                    bf16_t* __restrict__ x) {
    const size_t total = (size_t)MWORD * EDIM;
    for (size_t i = blockIdx.x * (size_t)blockDim.x + threadIdx.x; i < total;
         i += (size_t)gridDim.x * blockDim.x) {
        size_t row = i >> 8;          // EDIM = 256
        int    col = (int)(i & 255);
        int    id  = ids[row];
        x[i] = (bf16_t)emb[(size_t)id * EDIM + col];
    }
}

// masked word lengths, exclusive cumsum per conversation, w2_len output
__global__ void meta_kernel(const int* __restrict__ u_len,
                            const int* __restrict__ w_len,
                            int* __restrict__ wl, int* __restrict__ off,
                            float* __restrict__ w2len_out) {
    int b = threadIdx.x;
    if (b < BB) {
        int ul = u_len[b];
        int run = 0;
        for (int u = 0; u < UU; ++u) {
            int l = (u < ul) ? w_len[b * UU + u] : 0;
            wl[b * UU + u]  = l;
            off[b * UU + u] = run;
            run += l;
        }
        w2len_out[b] = (float)run;
    }
}

// ---------------- WMMA input-projection GEMM ----------------
// C[M x 768] = A_bf16[M x K] @ Wt_bf16[768 x K]^T + bias_f32[768]
// grid: (M/16, 768/128); block 256 = 8 waves, wave w -> N-tile blockIdx.y*128 + w*16
__global__ __launch_bounds__(256)
void gemm_bf16_kernel(const bf16_t* __restrict__ A,
                      const bf16_t* __restrict__ Wt,
                      const float* __restrict__ bias,
                      float* __restrict__ C, int M, int K) {
    const int lane = threadIdx.x & 31;
    const int wave = threadIdx.x >> 5;
    const int m0 = blockIdx.x * 16;
    const int n0 = blockIdx.y * 128 + wave * 16;

    v8f acc = {0.f, 0.f, 0.f, 0.f, 0.f, 0.f, 0.f, 0.f};
    const bf16_t* Abase = A  + (size_t)m0 * K;
    const bf16_t* Bbase = Wt + (size_t)n0 * K;

    for (int k0 = 0; k0 < K; k0 += 32) {
        v16bf a = load_frag(Abase, K, k0, lane);
        v16bf b = load_frag(Bbase, K, k0, lane);
        acc = wmma_bf16(a, b, acc);
    }

    const int n = n0 + (lane & 15);
    const int g = lane >> 4;
    const float bv = bias[n];
#pragma unroll
    for (int i = 0; i < 8; ++i) {
        int row = m0 + i + 8 * g;
        C[(size_t)row * G3 + n] = acc[i] + bv;
    }
}

// ---------------- GRU recurrence (both directions via blockIdx.y) ----------------
// One block handles 16 sequences of one direction. Hidden state lives in LDS
// (f32 master + bf16 WMMA operand copy). The time-invariant Whh r/z gate rows
// (512x256 bf16, 256KB) are cached in LDS once per block; the n-gate rows are
// streamed from L2 (they stay resident there). Each step: gh = h @ Whh^T via
// WMMA, gate math in registers, masked update, out = h*m to global.
__global__ __launch_bounds__(256)
void gru_rec_kernel(const float* __restrict__ xg_f, const float* __restrict__ xg_b,
                    const bf16_t* __restrict__ Wh_f, const bf16_t* __restrict__ Wh_b,
                    const float* __restrict__ bh_f, const float* __restrict__ bh_b,
                    const int* __restrict__ len,
                    float* __restrict__ out,       // ((seq*T)+t)*512 + dir*256 + col
                    bf16_t* hT_out,                // seq*512 + dir*256 + col (or null)
                    int T, int n_active) {
    const int dir = blockIdx.y;
    const int seq_base = blockIdx.x * 16;
    const float* __restrict__ xg = dir ? xg_b : xg_f;
    const bf16_t* __restrict__ Wh = dir ? Wh_b : Wh_f;
    const float* __restrict__ bh = dir ? bh_b : bh_f;
    float* __restrict__ outp = out + dir * HDIM;

    const int lane = threadIdx.x & 31;
    const int wave = threadIdx.x >> 5;
    const int g = lane >> 4;
    const int nn = lane & 15;

    extern __shared__ __align__(16) char smem[];
    float*  hF  = (float*)smem;
    bf16_t* hB  = (bf16_t*)(smem + REC_HF_BYTES);
    bf16_t* WhL = (bf16_t*)(smem + REC_HF_BYTES + REC_HB_BYTES);

    // init hidden state
    for (int i = threadIdx.x; i < 16 * HDIM; i += blockDim.x) {
        hF[i] = 0.0f;
        hB[i] = (bf16_t)0.0f;
    }
    // cache Whh rows [0, 512) (r and z gates) into LDS, 16B vectors
    {
        const v8bf* src = (const v8bf*)Wh;
        v8bf* dst = (v8bf*)WhL;
        const int nvec = REC_WCACHE_ROWS * HDIM / 8;
        for (int i = threadIdx.x; i < nvec; i += blockDim.x)
            dst[i] = src[i];
    }
    __syncthreads();

    // hoist time-invariant per-thread values
    int lenv[8];
#pragma unroll
    for (int i = 0; i < 8; ++i) {
        const int rm = i + 8 * g;
        lenv[i] = (rm < n_active) ? len[seq_base + rm] : 0;
    }
    float bhr[2], bhz[2], bhn[2];
#pragma unroll
    for (int ci = 0; ci < 2; ++ci) {
        const int ccol = wave * 16 + ci * 128 + nn;
        bhr[ci] = bh[ccol];
        bhz[ci] = bh[HDIM + ccol];
        bhn[ci] = bh[2 * HDIM + ccol];
    }

    for (int s = 0; s < T; ++s) {
        const int t = dir ? (T - 1 - s) : s;

        // gh = h @ Wh^T for this wave's 6 tiles: gates {r,z,n} x chunks {w*16, w*16+128}
        v8f acc[3][2];
#pragma unroll
        for (int gi = 0; gi < 3; ++gi)
#pragma unroll
            for (int ci = 0; ci < 2; ++ci)
                acc[gi][ci] = (v8f){0.f, 0.f, 0.f, 0.f, 0.f, 0.f, 0.f, 0.f};

        for (int k0 = 0; k0 < HDIM; k0 += 32) {
            v16bf a = load_frag(hB, HDIM, k0, lane);
#pragma unroll
            for (int gi = 0; gi < 3; ++gi)
#pragma unroll
                for (int ci = 0; ci < 2; ++ci) {
                    const int f0 = gi * HDIM + wave * 16 + ci * 128;
                    const bf16_t* bp = (gi < 2) ? (WhL + (size_t)f0 * HDIM)
                                                : (Wh + (size_t)f0 * HDIM);
                    v16bf b = load_frag(bp, HDIM, k0, lane);
                    acc[gi][ci] = wmma_bf16(a, b, acc[gi][ci]);
                }
        }
        __syncthreads();   // all waves finished reading hB

        // gate math + masked state update
#pragma unroll
        for (int ci = 0; ci < 2; ++ci) {
            const int ccol = wave * 16 + ci * 128 + nn;   // hidden column 0..255
#pragma unroll
            for (int i = 0; i < 8; ++i) {
                const int rm = i + 8 * g;
                const float hold = hF[rm * HDIM + ccol];
                float hnew = hold;
                if (rm < n_active) {
                    const int seq = seq_base + rm;
                    const size_t gx = ((size_t)seq * T + t) * G3;
                    const float rv = sigf(xg[gx + ccol] + acc[0][ci][i] + bhr[ci]);
                    const float zv =
                        sigf(xg[gx + HDIM + ccol] + acc[1][ci][i] + bhz[ci]);
                    const float nv = tanhf(xg[gx + 2 * HDIM + ccol] +
                                           rv * (acc[2][ci][i] + bhn[ci]));
                    const float hn = (1.0f - zv) * nv + zv * hold;
                    const bool m = (t < lenv[i]);
                    hnew = m ? hn : hold;
                    outp[((size_t)seq * T + t) * (2 * HDIM) + ccol] =
                        m ? hnew : 0.0f;
                }
                hF[rm * HDIM + ccol] = hnew;
                hB[rm * HDIM + ccol] = (bf16_t)hnew;
            }
        }
        __syncthreads();   // state update visible before next step's WMMA
    }

    if (hT_out) {
#pragma unroll
        for (int ci = 0; ci < 2; ++ci) {
            const int ccol = wave * 16 + ci * 128 + nn;
#pragma unroll
            for (int i = 0; i < 8; ++i) {
                const int rm = i + 8 * g;
                if (rm < n_active) {
                    const int seq = seq_base + rm;
                    hT_out[(size_t)seq * (2 * HDIM) + dir * HDIM + ccol] =
                        (bf16_t)hF[rm * HDIM + ccol];
                }
            }
        }
    }
}

// ---------------- scatter word outputs into compacted buffer ----------------
__global__ void scatter_kernel(const float* __restrict__ w_out,
                               const int* __restrict__ wl,
                               const int* __restrict__ off,
                               float* __restrict__ w2) {
    const int su = blockIdx.x;       // b*UU + u
    const int b = su / UU;
    const int l = wl[su];
    const int o = off[su];
    for (int w = 0; w < l; ++w) {
        const size_t src = ((size_t)su * WW + w) * (2 * HDIM);
        const size_t dst = ((size_t)b * (UU * WW) + o + w) * (2 * HDIM);
        for (int c = threadIdx.x; c < 2 * HDIM; c += blockDim.x)
            w2[dst + c] = w_out[src + c];
    }
}

// ---------------- host launcher ----------------
extern "C" void kernel_launch(void* const* d_in, const int* in_sizes, int n_in,
                              void* d_out, int out_size, void* d_ws, size_t ws_size,
                              hipStream_t stream) {
    (void)in_sizes; (void)n_in; (void)out_size; (void)ws_size;

    const int*   input_ids = (const int*)d_in[0];
    const int*   u_len     = (const int*)d_in[1];
    const int*   w_len     = (const int*)d_in[2];
    const float* emb       = (const float*)d_in[3];
    const float* wWif = (const float*)d_in[4],  * wWhf = (const float*)d_in[5];
    const float* wbif = (const float*)d_in[6],  * wbhf = (const float*)d_in[7];
    const float* wWib = (const float*)d_in[8],  * wWhb = (const float*)d_in[9];
    const float* wbib = (const float*)d_in[10], * wbhb = (const float*)d_in[11];
    const float* uWif = (const float*)d_in[12], * uWhf = (const float*)d_in[13];
    const float* ubif = (const float*)d_in[14], * ubhf = (const float*)d_in[15];
    const float* uWib = (const float*)d_in[16], * uWhb = (const float*)d_in[17];
    const float* ubib = (const float*)d_in[18], * ubhb = (const float*)d_in[19];

    float* out = (float*)d_out;
    float* u_output = out;                         // 8*40*512
    float* w2_out   = out + U_OUT_ELEMS;           // 8*3200*512
    float* w2_len   = out + U_OUT_ELEMS + W2_ELEMS;// 8

    char* ws = (char*)d_ws;
    size_t o = 0;
    auto alloc = [&](size_t bytes) -> char* {
        o = (o + 255) & ~(size_t)255;
        char* p = ws + o;
        o += bytes;
        return p;
    };

    bf16_t* wWif_b = (bf16_t*)alloc((size_t)G3 * EDIM * 2);
    bf16_t* wWib_b = (bf16_t*)alloc((size_t)G3 * EDIM * 2);
    bf16_t* wWhf_b = (bf16_t*)alloc((size_t)G3 * HDIM * 2);
    bf16_t* wWhb_b = (bf16_t*)alloc((size_t)G3 * HDIM * 2);
    bf16_t* uWif_b = (bf16_t*)alloc((size_t)G3 * 2 * HDIM * 2);
    bf16_t* uWib_b = (bf16_t*)alloc((size_t)G3 * 2 * HDIM * 2);
    bf16_t* uWhf_b = (bf16_t*)alloc((size_t)G3 * HDIM * 2);
    bf16_t* uWhb_b = (bf16_t*)alloc((size_t)G3 * HDIM * 2);
    bf16_t* x_b    = (bf16_t*)alloc((size_t)MWORD * EDIM * 2);
    float*  xgf    = (float*) alloc((size_t)MWORD * G3 * 4);
    float*  xgb    = (float*) alloc((size_t)MWORD * G3 * 4);
    float*  w_outb = (float*) alloc((size_t)MWORD * 2 * HDIM * 4);
    bf16_t* utt_in = (bf16_t*)alloc((size_t)NSEQ * 2 * HDIM * 2);
    float*  ugf    = (float*) alloc((size_t)NSEQ * G3 * 4);
    float*  ugb    = (float*) alloc((size_t)NSEQ * G3 * 4);
    int*    wl     = (int*)   alloc((size_t)NSEQ * 4);
    int*    offb   = (int*)   alloc((size_t)NSEQ * 4);

    // 1) pack weights to bf16
    f32_to_bf16_kernel<<<512, 256, 0, stream>>>(wWif, wWif_b, G3 * EDIM);
    f32_to_bf16_kernel<<<512, 256, 0, stream>>>(wWib, wWib_b, G3 * EDIM);
    f32_to_bf16_kernel<<<512, 256, 0, stream>>>(wWhf, wWhf_b, G3 * HDIM);
    f32_to_bf16_kernel<<<512, 256, 0, stream>>>(wWhb, wWhb_b, G3 * HDIM);
    f32_to_bf16_kernel<<<512, 256, 0, stream>>>(uWif, uWif_b, G3 * 2 * HDIM);
    f32_to_bf16_kernel<<<512, 256, 0, stream>>>(uWib, uWib_b, G3 * 2 * HDIM);
    f32_to_bf16_kernel<<<512, 256, 0, stream>>>(uWhf, uWhf_b, G3 * HDIM);
    f32_to_bf16_kernel<<<512, 256, 0, stream>>>(uWhb, uWhb_b, G3 * HDIM);

    // 2) embedding gather + lengths/offsets
    embed_gather_kernel<<<2048, 256, 0, stream>>>(input_ids, emb, x_b);
    meta_kernel<<<1, 32, 0, stream>>>(u_len, w_len, wl, offb, w2_len);

    // 3) word-level input projections (WMMA)
    gemm_bf16_kernel<<<dim3(MWORD / 16, 6), 256, 0, stream>>>(x_b, wWif_b, wbif,
                                                              xgf, MWORD, EDIM);
    gemm_bf16_kernel<<<dim3(MWORD / 16, 6), 256, 0, stream>>>(x_b, wWib_b, wbib,
                                                              xgb, MWORD, EDIM);

    // 4) word-level recurrence (both dirs), writes w_out + utt_input=concat(hf,hb)
    gru_rec_kernel<<<dim3(NSEQ / 16, 2), 256, REC_SMEM_BYTES, stream>>>(
        xgf, xgb, wWhf_b, wWhb_b, wbhf, wbhb, wl, w_outb, utt_in, WW, 16);

    // 5) utterance-level input projections (WMMA)
    gemm_bf16_kernel<<<dim3(NSEQ / 16, 6), 256, 0, stream>>>(utt_in, uWif_b, ubif,
                                                             ugf, NSEQ, 2 * HDIM);
    gemm_bf16_kernel<<<dim3(NSEQ / 16, 6), 256, 0, stream>>>(utt_in, uWib_b, ubib,
                                                             ugb, NSEQ, 2 * HDIM);

    // 6) utterance-level recurrence writes u_output directly
    gru_rec_kernel<<<dim3(1, 2), 256, REC_SMEM_BYTES, stream>>>(
        ugf, ugb, uWhf_b, uWhb_b, ubhf, ubhb, u_len, u_output, (bf16_t*)nullptr,
        UU, BB);

    // 7) compact word outputs
    zero_f32_kernel<<<4096, 256, 0, stream>>>(w2_out, W2_ELEMS);
    scatter_kernel<<<NSEQ, 256, 0, stream>>>(w_outb, wl, offb, w2_out);
}
